// CausalSelfAttention_7859790152104
// MI455X (gfx1250) — matrix operand order repaired and checked by reading, not compiled, not run on hardware
//
#include <hip/hip_runtime.h>
#include <hip/hip_bf16.h>
#include <stdint.h>

typedef __attribute__((ext_vector_type(16))) __bf16 v16bf;
typedef __attribute__((ext_vector_type(8)))  float  v8f;

#define T_SEQ 4096
#define C_EMB 768
#define N_HEAD 12
#define HEAD_DIM 64
#define THREE_C 2304

// native RNE conversion; backend may use v_cvt_pk_bf16_f32-class ops
static __device__ __forceinline__ unsigned short f2bf(float f) {
    union { __bf16 b; unsigned short u; } c;
    c.b = (__bf16)f;
    return c.u;
}
static __device__ __forceinline__ unsigned int pack2bf(float a, float b) {
    union { unsigned short u[2]; unsigned int w; } c;
    c.u[0] = f2bf(a); c.u[1] = f2bf(b);
    return c.w;
}

// exchange value with partner lane in the other 16-lane half (lane L <-> L^16)
static __device__ __forceinline__ unsigned int xhalf_u32(unsigned int v) {
#if __has_builtin(__builtin_amdgcn_permlanex16)
    return __builtin_amdgcn_permlanex16(v, v, 0x76543210u, 0xFEDCBA98u, false, false);
#else
    return __shfl_xor((int)v, 16, 32);
#endif
}
static __device__ __forceinline__ float xhalf_f32(float v) {
    union { float f; unsigned u; } x; x.f = v;
    x.u = xhalf_u32(x.u);
    return x.f;
}

union FragAB { v16bf v; uint4 q[2]; unsigned int w[8]; unsigned short h[16]; };

// ---------------- prep kernels ----------------
__global__ void k_f2bf(const float* __restrict__ src, unsigned short* __restrict__ dst, int n) {
    int i = blockIdx.x * blockDim.x + threadIdx.x;
    if (i < n) dst[i] = f2bf(src[i]);
}

// src is K x N row-major fp32; dst is N x K row-major bf16 (i.e. transposed)
__global__ void k_transpose_bf(const float* __restrict__ src, unsigned short* __restrict__ dst,
                               int K, int N) {
    int i = blockIdx.x * blockDim.x + threadIdx.x;
    if (i >= K * N) return;
    int n = i % N;
    int k = i / N;
    dst[(size_t)n * K + k] = f2bf(src[(size_t)k * N + n]);
}

// ---------------- GEMM: C(MxN) = A(MxK) * BT(NxK)^T + bias ----------------
// WG tile 64x256, 8 waves as 2(M)x4(N); wave tile 32x64 = 2 A-frags x 4 B-frags.
// Global->register prefetch pipelined against WMMA compute.
// MODE 0: qkv projection -> scatter into Qb (scaled 1/8), Kb (H,T,D), Vt (H,D,T), bf16
// MODE 1: out projection -> fp32 out (T x 768) + bias
template <int MODE>
__global__ __launch_bounds__(256) void k_gemm(const unsigned short* __restrict__ A,
                                              const unsigned short* __restrict__ BT,
                                              const float* __restrict__ bias,
                                              int Kdim, int Nsz,
                                              unsigned short* __restrict__ Qb,
                                              unsigned short* __restrict__ Kb,
                                              unsigned short* __restrict__ Vt,
                                              float* __restrict__ outF) {
    __shared__ unsigned short Ash[64 * 40];     // 64 rows x 32 (pad->40)
    __shared__ unsigned short Bsh[256 * 40];    // 256 rows x 32 (pad->40)

    const int tid  = threadIdx.x;
    const int lane = tid & 31;
    const int wave = tid >> 5;
    const int wM   = wave & 1;           // 2 M-waves (32 rows each)
    const int wN   = wave >> 1;          // 4 N-waves (64 cols each)
    const int half = lane >> 4;
    const int lr   = lane & 15;
    const int m0   = blockIdx.x * 64;
    const int n0   = blockIdx.y * 256;

    // per-thread staging addresses
    const int arow = tid >> 2, acol = (tid & 3) * 8;        // A: 64x32, 1 x uint4
    const int brow = tid;                                   // B: 256x32, 4 x uint4 (full row)
    const unsigned short* aptr = &A[(size_t)(m0 + arow) * Kdim + acol];
    const unsigned short* bptr = &BT[(size_t)(n0 + brow) * Kdim];

    // prefetch k-step 0
    uint4 pa  = *(const uint4*)aptr;
    uint4 pb0 = *(const uint4*)(bptr + 0);
    uint4 pb1 = *(const uint4*)(bptr + 8);
    uint4 pb2 = *(const uint4*)(bptr + 16);
    uint4 pb3 = *(const uint4*)(bptr + 24);

    v8f zero = {};
    v8f acc[2][4];
#pragma unroll
    for (int ai = 0; ai < 2; ++ai)
#pragma unroll
        for (int j = 0; j < 4; ++j) acc[ai][j] = zero;

    for (int k0 = 0; k0 < Kdim; k0 += 32) {
        __syncthreads();                       // previous iteration finished reading LDS
        *(uint4*)&Ash[arow * 40 + acol] = pa;
        *(uint4*)&Bsh[brow * 40 + 0]  = pb0;
        *(uint4*)&Bsh[brow * 40 + 8]  = pb1;
        *(uint4*)&Bsh[brow * 40 + 16] = pb2;
        *(uint4*)&Bsh[brow * 40 + 24] = pb3;
        __syncthreads();

        const int kn = k0 + 32;
        if (kn < Kdim) {                       // prefetch next k-step (overlaps WMMAs below)
            pa  = *(const uint4*)(aptr + kn);
            pb0 = *(const uint4*)(bptr + kn + 0);
            pb1 = *(const uint4*)(bptr + kn + 8);
            pb2 = *(const uint4*)(bptr + kn + 16);
            pb3 = *(const uint4*)(bptr + kn + 24);
        }

        // batch all fragment loads, then batch the WMMAs
        FragAB a[2], b[4];
#pragma unroll
        for (int ai = 0; ai < 2; ++ai) {
            int row = wM * 32 + ai * 16 + lr;
            a[ai].q[0] = *(const uint4*)&Ash[row * 40 + half * 8];
            a[ai].q[1] = *(const uint4*)&Ash[row * 40 + 16 + half * 8];
        }
#pragma unroll
        for (int j = 0; j < 4; ++j) {
            int row = wN * 64 + j * 16 + lr;
            b[j].q[0] = *(const uint4*)&Bsh[row * 40 + half * 8];
            b[j].q[1] = *(const uint4*)&Bsh[row * 40 + 16 + half * 8];
        }
#pragma unroll
        for (int ai = 0; ai < 2; ++ai)
#pragma unroll
            for (int j = 0; j < 4; ++j)
                acc[ai][j] = __builtin_amdgcn_wmma_f32_16x16x32_bf16(
                    false, a[ai].v, false, b[j].v, (short)0, acc[ai][j], false, false);
    }

    // epilogue
#pragma unroll
    for (int ai = 0; ai < 2; ++ai)
#pragma unroll
        for (int j = 0; j < 4; ++j) {
            int nn = n0 + wN * 64 + j * 16 + lr;
            float bv = bias[nn];
#pragma unroll
            for (int r = 0; r < 8; ++r) {
                int tt = m0 + wM * 32 + ai * 16 + r + half * 8;
                float val = acc[ai][j][r] + bv;
                if (MODE == 0) {
                    int sec = nn / C_EMB;
                    int rem = nn - sec * C_EMB;
                    int hh = rem >> 6, dd = rem & 63;
                    size_t idx = ((size_t)hh * T_SEQ + tt) * HEAD_DIM + dd;
                    if (sec == 0)      Qb[idx] = f2bf(val * 0.125f);   // fold 1/sqrt(64)
                    else if (sec == 1) Kb[idx] = f2bf(val);
                    else Vt[((size_t)hh * HEAD_DIM + dd) * T_SEQ + tt] = f2bf(val);
                } else {
                    outF[(size_t)tt * Nsz + nn] = val;
                }
            }
        }
}

// ---------------- flash attention (transposed-S formulation) ----------------
// S^T = K Q^T  -> query lives in the lane; softmax stats are per-lane scalars.
// O^T = V^T P^T; P^T C-frag -> B-frag via 4 permlanex16. K/V staging prefetched
// into registers one block ahead so global latency overlaps WMMA + softmax.
// grid: (T/64, H); block: 128 (4 waves, each owns a 16-row query tile)
__global__ __launch_bounds__(128) void k_attn(const unsigned short* __restrict__ Qb,
                                              const unsigned short* __restrict__ Kb,
                                              const unsigned short* __restrict__ Vt,
                                              unsigned short* __restrict__ attnb) {
    __shared__ unsigned short Ksh[32 * 72];        // 32 keys x 64 d (pad->72)
    __shared__ unsigned short Vsh[64 * 40];        // 64 d x 32 keys (pad->40)

    const int tid  = threadIdx.x;
    const int lane = tid & 31;
    const int wave = tid >> 5;
    const int half = lane >> 4;
    const int lr   = lane & 15;
    const int h    = blockIdx.y;
    const int m0   = blockIdx.x * 64;
    const int q0   = m0 + wave * 16;
    const int q0u  = __builtin_amdgcn_readfirstlane(q0);   // wave-uniform scalar

    // Q^T B-fragments (K-dim = d), loaded once: lane = query, d chunks at 8*half and 16+8*half (table order)
    FragAB qb[2];
    const unsigned short* qrow = Qb + ((size_t)h * T_SEQ + q0 + lr) * HEAD_DIM;
#pragma unroll
    for (int ks = 0; ks < 2; ++ks) {
        qb[ks].q[0] = *(const uint4*)(qrow + ks * 32 + half * 8);
        qb[ks].q[1] = *(const uint4*)(qrow + ks * 32 + 16 + half * 8);
    }

    v8f zero = {};
    v8f acc[4];                          // O^T tiles: d = j*16 + r + 8*half, query = lr
#pragma unroll
    for (int j = 0; j < 4; ++j) acc[j] = zero;
    float mrun = -3.0e38f, lrun = 0.0f;  // per-lane (per-query) softmax state

    const int nb = (m0 + 64) >> 5;       // 32-key blocks up to WG diagonal

    // per-thread staging addresses
    const int krow = tid >> 2, kcol = (tid & 3) * 16;    // K: 32x64
    const int vrow = tid >> 1, vcol = (tid & 1) * 16;    // V^T: 64x32
    const unsigned short* kbase = Kb + ((size_t)h * T_SEQ + krow) * HEAD_DIM + kcol;
    const unsigned short* vbase = Vt + ((size_t)h * HEAD_DIM + vrow) * T_SEQ + vcol;

    // prefetch block 0
    uint4 pk0 = *(const uint4*)(kbase + 0);
    uint4 pk1 = *(const uint4*)(kbase + 8);
    uint4 pv0 = *(const uint4*)(vbase + 0);
    uint4 pv1 = *(const uint4*)(vbase + 8);

    for (int b = 0; b < nb; ++b) {
        const int kb = b * 32;
        __syncthreads();                 // previous block finished reading LDS
        *(uint4*)&Ksh[krow * 72 + kcol]     = pk0;
        *(uint4*)&Ksh[krow * 72 + kcol + 8] = pk1;
        *(uint4*)&Vsh[vrow * 40 + vcol]     = pv0;
        *(uint4*)&Vsh[vrow * 40 + vcol + 8] = pv1;
        __syncthreads();

        if (b + 1 < nb) {                // prefetch next block (overlaps compute below)
            const size_t koff = (size_t)(kb + 32) * HEAD_DIM;   // advance 32 key rows
            pk0 = *(const uint4*)(kbase + koff);
            pk1 = *(const uint4*)(kbase + koff + 8);
            pv0 = *(const uint4*)(vbase + kb + 32);             // advance 32 key cols
            pv1 = *(const uint4*)(vbase + kb + 32 + 8);
        }

        if (kb < q0u + 16) {   // scalar-uniform: block touches this wave's causal region
            // batch all LDS fragment loads for this block
            FragAB ka[2][2], va[4];
#pragma unroll
            for (int t = 0; t < 2; ++t) {
                int row = t * 16 + lr;
#pragma unroll
                for (int ks = 0; ks < 2; ++ks) {
                    ka[t][ks].q[0] = *(const uint4*)&Ksh[row * 72 + ks * 32 + half * 8];
                    ka[t][ks].q[1] = *(const uint4*)&Ksh[row * 72 + ks * 32 + 16 + half * 8];
                }
            }
#pragma unroll
            for (int j = 0; j < 4; ++j) {
                int row = j * 16 + lr;
                va[j].q[0] = *(const uint4*)&Vsh[row * 40 + half * 16];
                va[j].q[1] = *(const uint4*)&Vsh[row * 40 + half * 16 + 8];
            }

            // S^T : two 16(key)x16(query) tiles
            v8f st[2];
            st[0] = zero; st[1] = zero;
#pragma unroll
            for (int t = 0; t < 2; ++t)
#pragma unroll
                for (int ks = 0; ks < 2; ++ks)
                    st[t] = __builtin_amdgcn_wmma_f32_16x16x32_bf16(
                        false, ka[t][ks].v, false, qb[ks].v, (short)0, st[t], false, false);

            // causal mask: only diagonal block(s) need per-element masking
            if (kb + 31 > q0u) {
                const int qg = q0 + lr;
#pragma unroll
                for (int t = 0; t < 2; ++t)
#pragma unroll
                    for (int r = 0; r < 8; ++r) {
                        int kg = kb + t * 16 + half * 8 + r;
                        if (kg > qg) st[t][r] = -3.0e38f;
                    }
            }

            // per-query softmax: in-lane reduce + one cross-half exchange
            float mx = st[0][0];
#pragma unroll
            for (int r = 1; r < 8; ++r) mx = fmaxf(mx, st[0][r]);
#pragma unroll
            for (int r = 0; r < 8; ++r) mx = fmaxf(mx, st[1][r]);
            mx = fmaxf(mx, xhalf_f32(mx));
            const float mnew = fmaxf(mrun, mx);
            const float sc = __expf(mrun - mnew);

            float p0[8], p1[8];
            float sum = 0.0f;
#pragma unroll
            for (int r = 0; r < 8; ++r) { p0[r] = __expf(st[0][r] - mnew); sum += p0[r]; }
#pragma unroll
            for (int r = 0; r < 8; ++r) { p1[r] = __expf(st[1][r] - mnew); sum += p1[r]; }
            sum += xhalf_f32(sum);
            mrun = mnew;
            lrun = lrun * sc + sum;

            // rescale O^T accumulators (per-lane scalar, no shuffles)
#pragma unroll
            for (int j = 0; j < 4; ++j)
#pragma unroll
                for (int r = 0; r < 8; ++r) acc[j][r] *= sc;

            // build P^T B-fragment in registers (4 permlanex16)
            unsigned int packA[4], packB[4];
#pragma unroll
            for (int i = 0; i < 4; ++i) {
                packA[i] = pack2bf(p0[2 * i], p0[2 * i + 1]);
                packB[i] = pack2bf(p1[2 * i], p1[2 * i + 1]);
            }
            FragAB pb;
#pragma unroll
            for (int i = 0; i < 4; ++i) {
                unsigned int src = half ? packA[i] : packB[i];   // send what partner needs
                unsigned int x   = xhalf_u32(src);               // receive what we need
                pb.w[i]     = half ? x : packA[i];
                pb.w[4 + i] = half ? packB[i] : x;
            }

            // O^T += V^T P^T (fragments already in registers)
#pragma unroll
            for (int j = 0; j < 4; ++j)
                acc[j] = __builtin_amdgcn_wmma_f32_16x16x32_bf16(
                    false, va[j].v, false, pb.v, (short)0, acc[j], false, false);
        }
    }

    // normalize (per-lane) and store bf16 attention output (T x 768)
    const float inv = (lrun > 0.0f) ? (1.0f / lrun) : 0.0f;
    const size_t trow = (size_t)(q0 + lr) * C_EMB + h * HEAD_DIM;
#pragma unroll
    for (int j = 0; j < 4; ++j)
#pragma unroll
        for (int r = 0; r < 8; ++r)
            attnb[trow + j * 16 + r + 8 * half] = f2bf(acc[j][r] * inv);
}

// ---------------- launch ----------------
extern "C" void kernel_launch(void* const* d_in, const int* in_sizes, int n_in,
                              void* d_out, int out_size, void* d_ws, size_t ws_size,
                              hipStream_t stream) {
    const float* x     = (const float*)d_in[0];
    const float* W_qkv = (const float*)d_in[1];
    const float* b_qkv = (const float*)d_in[2];
    const float* W_out = (const float*)d_in[3];
    const float* b_out = (const float*)d_in[4];
    float* out = (float*)d_out;

    char* ws = (char*)d_ws;
    size_t off = 0;
    auto carve = [&](size_t bytes) -> char* {
        char* p = ws + off;
        off = (off + bytes + 255) & ~(size_t)255;
        return p;
    };
    unsigned short* xb     = (unsigned short*)carve((size_t)T_SEQ * C_EMB * 2);
    unsigned short* wqkvT  = (unsigned short*)carve((size_t)THREE_C * C_EMB * 2);
    unsigned short* woutT  = (unsigned short*)carve((size_t)C_EMB * C_EMB * 2);
    unsigned short* Qb     = (unsigned short*)carve((size_t)N_HEAD * T_SEQ * HEAD_DIM * 2);
    unsigned short* Kb     = (unsigned short*)carve((size_t)N_HEAD * T_SEQ * HEAD_DIM * 2);
    unsigned short* Vt     = (unsigned short*)carve((size_t)N_HEAD * T_SEQ * HEAD_DIM * 2);
    unsigned short* attnb  = (unsigned short*)carve((size_t)T_SEQ * C_EMB * 2);
    (void)ws_size; (void)in_sizes; (void)n_in; (void)out_size;

    k_f2bf<<<(T_SEQ * C_EMB) / 256, 256, 0, stream>>>(x, xb, T_SEQ * C_EMB);
    k_transpose_bf<<<(C_EMB * THREE_C) / 256, 256, 0, stream>>>(W_qkv, wqkvT, C_EMB, THREE_C);
    k_transpose_bf<<<(C_EMB * C_EMB) / 256, 256, 0, stream>>>(W_out, woutT, C_EMB, C_EMB);

    k_gemm<0><<<dim3(T_SEQ / 64, THREE_C / 256), 256, 0, stream>>>(
        xb, wqkvT, b_qkv, C_EMB, THREE_C, Qb, Kb, Vt, nullptr);

    k_attn<<<dim3(T_SEQ / 64, N_HEAD), 128, 0, stream>>>(Qb, Kb, Vt, attnb);

    k_gemm<1><<<dim3(T_SEQ / 64, C_EMB / 256), 256, 0, stream>>>(
        attnb, woutT, b_out, C_EMB, C_EMB, nullptr, nullptr, nullptr, out);
}
